// VILSTM_11845519802777
// MI455X (gfx1250) — compile-verified, matrix operand order checked by reading, not run on hardware
//
#include <hip/hip_runtime.h>
#include <hip/hip_bf16.h>

// ---------------- CDNA5 WMMA types ----------------
typedef __attribute__((ext_vector_type(16))) __bf16 v16bf;
typedef __attribute__((ext_vector_type(8)))  float  v8f;

#define TT 64
#define BB 256
#define DD 1024
#define HH 1024
#define FF 1024
#define LL 2
#define G4 4096   // 4*H

// scheduler group pinning (composable-kernel style IGLP hints)
#if defined(__has_builtin)
#  if __has_builtin(__builtin_amdgcn_sched_group_barrier)
#    define SCHED_GROUP(mask, size, id) __builtin_amdgcn_sched_group_barrier(mask, size, id)
#  else
#    define SCHED_GROUP(mask, size, id)
#  endif
#else
#  define SCHED_GROUP(mask, size, id)
#endif
#define SG_WMMA      0x008   // MFMA/WMMA class
#define SG_VMEM_READ 0x020   // VMEM read class

// fp32 -> bf16 round-to-nearest-even (bit trick; no __bf16 arithmetic)
__device__ __forceinline__ unsigned short f2bf(float f) {
    union { float f; unsigned u; } v; v.f = f;
    unsigned r = (v.u + 0x7FFFu + ((v.u >> 16) & 1u)) >> 16;
    return (unsigned short)r;
}
__device__ __forceinline__ unsigned pack_bf2(float lo, float hi) {
    return (unsigned)f2bf(lo) | ((unsigned)f2bf(hi) << 16);
}

// Load one 16x32 bf16 fragment with a lane-adjusted base pointer (already
// includes row*stride + 8*half) and a compile-time K element offset. Both
// 16-byte loads get constant immediate offsets (2k and 2k+32 bytes).
// CDNA5 16-bit layout: lane = (row/col)%16; lanes 0-15 hold K=[0..7],[16..23],
// lanes 16-31 hold K=[8..15],[24..31] (the +8*half in the base).
__device__ __forceinline__ v16bf load_frag_k(const unsigned short* __restrict__ pbase, int k) {
    union { v16bf v; uint4 q[2]; } u;
    u.q[0] = *reinterpret_cast<const uint4*>(pbase + k);
    u.q[1] = *reinterpret_cast<const uint4*>(pbase + k + 16);
    return u.v;
}

__device__ __forceinline__ v8f wmma_bf16(v16bf a, v16bf b, v8f c) {
    // (neg_a, A, neg_b, B, c_mod, C, reuse_a, reuse_b)
    return __builtin_amdgcn_wmma_f32_16x16x32_bf16(false, a, false, b, (short)0, c, false, false);
}

__device__ __forceinline__ float sigmoidf_fast(float x) {
    return 1.0f / (1.0f + __expf(-x));
}

union F8 { float4 v[2]; float s[8]; };

// ---------------- helper kernels ----------------
__global__ void cvt_f32_bf16_kernel(const float4* __restrict__ in,
                                    ushort4* __restrict__ out, int n4) {
    int i = blockIdx.x * blockDim.x + threadIdx.x;
    int stride = gridDim.x * blockDim.x;
    for (; i < n4; i += stride) {
        float4 v = in[i];
        ushort4 o;
        o.x = f2bf(v.x); o.y = f2bf(v.y); o.z = f2bf(v.z); o.w = f2bf(v.w);
        out[i] = o;
    }
}

__global__ void zero_u32_kernel(unsigned* __restrict__ p, int n) {
    int i = blockIdx.x * blockDim.x + threadIdx.x;
    int stride = gridDim.x * blockDim.x;
    for (; i < n; i += stride) p[i] = 0u;
}

// ---------------- featb = features @ Wfh[l]^T + b[l]  (L,B,4H) fp32 ----------------
// block = 128 threads (4 waves); each wave: 16(M) x 64(N) tile, K = F = 1024.
// grid = (4H/64, B/64, L). Fully unrolled K, 2-chunk pipeline enforced by
// sched_group_barrier: [VMEM 10][VMEM 10] then per chunk [VMEM 10][WMMA 4].
__global__ __launch_bounds__(128)
void featb_gemm_kernel(const unsigned short* __restrict__ feat_bf, // (B,F) bf16
                       const unsigned short* __restrict__ Wfh_bf,  // (L,4H,F) bf16
                       const float* __restrict__ bias,             // (L,4H)
                       float* __restrict__ featb)                  // (L,B,4H)
{
    const int l    = blockIdx.z;
    const int wave = threadIdx.x >> 5;
    const int lane = threadIdx.x & 31;
    const int row  = lane & 15;
    const int half = lane >> 4;

    const int n0 = blockIdx.x * 64;
    const int m0 = blockIdx.y * 64 + wave * 16;

    const unsigned short* W  = Wfh_bf + (size_t)l * G4 * FF;
    const unsigned short* Ab = feat_bf + (size_t)(m0 + row) * FF + 8 * half;
    const unsigned short* Bb[4];
#pragma unroll
    for (int nt = 0; nt < 4; ++nt)
        Bb[nt] = W + (size_t)(n0 + nt * 16 + row) * FF + 8 * half;

    v8f acc[4] = {v8f{}, v8f{}, v8f{}, v8f{}};

    const int NC = FF / 32;   // 32 chunks
    v16bf aA, aB, bA[4], bB[4];
    aA = load_frag_k(Ab, 0);
#pragma unroll
    for (int nt = 0; nt < 4; ++nt) bA[nt] = load_frag_k(Bb[nt], 0);
    aB = load_frag_k(Ab, 32);
#pragma unroll
    for (int nt = 0; nt < 4; ++nt) bB[nt] = load_frag_k(Bb[nt], 32);
    SCHED_GROUP(SG_VMEM_READ, 10, 0);
    SCHED_GROUP(SG_VMEM_READ, 10, 0);

#pragma unroll
    for (int kc = 0; kc < NC - 2; ++kc) {
        const int ko = (kc + 2) * 32;
        v16bf aN = load_frag_k(Ab, ko);
        v16bf bN[4];
#pragma unroll
        for (int nt = 0; nt < 4; ++nt) bN[nt] = load_frag_k(Bb[nt], ko);
#pragma unroll
        for (int nt = 0; nt < 4; ++nt) acc[nt] = wmma_bf16(aA, bA[nt], acc[nt]);
        aA = aB;
#pragma unroll
        for (int nt = 0; nt < 4; ++nt) bA[nt] = bB[nt];
        aB = aN;
#pragma unroll
        for (int nt = 0; nt < 4; ++nt) bB[nt] = bN[nt];
        SCHED_GROUP(SG_VMEM_READ, 10, 0);
        SCHED_GROUP(SG_WMMA, 4, 0);
    }
#pragma unroll
    for (int nt = 0; nt < 4; ++nt) acc[nt] = wmma_bf16(aA, bA[nt], acc[nt]);
#pragma unroll
    for (int nt = 0; nt < 4; ++nt) acc[nt] = wmma_bf16(aB, bB[nt], acc[nt]);
    SCHED_GROUP(SG_WMMA, 4, 0);
    SCHED_GROUP(SG_WMMA, 4, 0);

    float* out = featb + (size_t)l * BB * G4;
#pragma unroll
    for (int nt = 0; nt < 4; ++nt) {
#pragma unroll
        for (int i = 0; i < 8; ++i) {
            int m = m0 + i + 8 * half;
            int n = n0 + nt * 16 + row;
            out[(size_t)m * G4 + n] = acc[nt][i] + bias[(size_t)l * G4 + n];
        }
    }
}

// ---------------- one LSTM timestep ----------------
// gates(B,4H) = x_t @ Wih^T + h_prev @ Whh^T + featb ; then activations + mask.
// block = 128 threads (4 waves); wave g computes gate g's 32(batch) x 64(hidden)
// tile: 2 A-frags x 4 B-frags -> 8 WMMA per 32-K chunk. 64 chunks fully
// unrolled; pipeline order pinned via sched_group_barrier:
// [VMEM 12][VMEM 12] then per chunk [VMEM 12][WMMA 8] -> 24 loads in flight.
// grid = (H/64, B/32).
__global__ __launch_bounds__(128)
void lstm_step_kernel(const unsigned short* __restrict__ xin_bf,   // (T,B,1024) bf16 layer input
                      const unsigned short* __restrict__ hprev_bf, // (B,H) bf16
                      unsigned short* __restrict__ hnext_bf,       // (B,H) bf16
                      const unsigned short* __restrict__ Wih_bf,   // (4H,D) bf16 (this layer)
                      const unsigned short* __restrict__ Whh_bf,   // (4H,H) bf16 (this layer)
                      const float* __restrict__ featb_l,           // (B,4H) fp32 (this layer)
                      const int*  __restrict__ length,             // (B,)
                      float* __restrict__ h_f32,                   // (B,H) state
                      float* __restrict__ c_f32,                   // (B,H) state
                      unsigned short* __restrict__ yout_bf,        // (T,B,H) or null
                      float* __restrict__ yout_f32,                // (T,B,H) or null (d_out)
                      float* __restrict__ hn_out,                  // (B,H) or null
                      float* __restrict__ cn_out,                  // (B,H) or null
                      int t)
{
    __shared__ float lds[4][32][64];   // 32 KB: gate-major tiles

    const int wave = threadIdx.x >> 5;   // 0..3 = gate i,f,g,o
    const int lane = threadIdx.x & 31;
    const int row  = lane & 15;
    const int half = lane >> 4;

    const int j0 = blockIdx.x * 64;      // hidden-column tile
    const int m0 = blockIdx.y * 32;      // batch-row tile (32 rows)

    // Lane-adjusted base pointers (computed once; all loads use imm offsets).
    const unsigned short* Ax0 = xin_bf + (size_t)t * BB * DD + (size_t)(m0 + row) * DD + 8 * half;
    const unsigned short* Ax1 = Ax0 + (size_t)16 * DD;
    const unsigned short* Ah0 = hprev_bf + (size_t)(m0 + row) * HH + 8 * half;
    const unsigned short* Ah1 = Ah0 + (size_t)16 * HH;

    const unsigned short* Bih[4];
    const unsigned short* Bhh[4];
#pragma unroll
    for (int nt = 0; nt < 4; ++nt) {
        int g = wave * HH + j0 + nt * 16 + row;   // gate row in (4H, K)
        Bih[nt] = Wih_bf + (size_t)g * DD + 8 * half;
        Bhh[nt] = Whh_bf + (size_t)g * HH + 8 * half;
    }

    // acc[r][nt]: r = 16-row sub-block, nt = 16-col sub-tile
    v8f acc[2][4] = {{v8f{}, v8f{}, v8f{}, v8f{}}, {v8f{}, v8f{}, v8f{}, v8f{}}};

    // ---- fully unrolled, 2-chunk-deep software-pipelined GEMM ----
    // chunk kc covers K elements [kc*32, kc*32+32); kc < 32 -> D segment (x/Wih),
    // kc >= 32 -> H segment (h/Whh). All selects fold at compile time.
    v16bf a0A, a1A, a0B, a1B, bA[4], bB[4];
    a0A = load_frag_k(Ax0, 0);
    a1A = load_frag_k(Ax1, 0);
#pragma unroll
    for (int nt = 0; nt < 4; ++nt) bA[nt] = load_frag_k(Bih[nt], 0);
    a0B = load_frag_k(Ax0, 32);
    a1B = load_frag_k(Ax1, 32);
#pragma unroll
    for (int nt = 0; nt < 4; ++nt) bB[nt] = load_frag_k(Bih[nt], 32);
    SCHED_GROUP(SG_VMEM_READ, 12, 0);
    SCHED_GROUP(SG_VMEM_READ, 12, 0);

#pragma unroll
    for (int kc = 0; kc < 62; ++kc) {
        // issue loads for chunk kc+2 (ahead of this chunk's WMMAs)
        const int kn = kc + 2;
        const unsigned short* A0b = (kn < 32) ? Ax0 : Ah0;
        const unsigned short* A1b = (kn < 32) ? Ax1 : Ah1;
        const int ko = ((kn < 32) ? kn : (kn - 32)) * 32;
        v16bf a0N = load_frag_k(A0b, ko);
        v16bf a1N = load_frag_k(A1b, ko);
        v16bf bN[4];
#pragma unroll
        for (int nt = 0; nt < 4; ++nt) {
            const unsigned short* Bn = (kn < 32) ? Bih[nt] : Bhh[nt];
            bN[nt] = load_frag_k(Bn, ko);
        }
        // consume stage A (chunk kc)
#pragma unroll
        for (int nt = 0; nt < 4; ++nt) {
            acc[0][nt] = wmma_bf16(a0A, bA[nt], acc[0][nt]);
            acc[1][nt] = wmma_bf16(a1A, bA[nt], acc[1][nt]);
        }
        // SSA stage shift (fully unrolled -> pure renames, no moves)
        a0A = a0B; a1A = a1B;
#pragma unroll
        for (int nt = 0; nt < 4; ++nt) bA[nt] = bB[nt];
        a0B = a0N; a1B = a1N;
#pragma unroll
        for (int nt = 0; nt < 4; ++nt) bB[nt] = bN[nt];
        // pin schedule: this chunk's 12 loads, then 8 WMMAs of chunk kc
        SCHED_GROUP(SG_VMEM_READ, 12, 0);
        SCHED_GROUP(SG_WMMA, 8, 0);
    }
    // epilogue: chunks 62, 63
#pragma unroll
    for (int nt = 0; nt < 4; ++nt) {
        acc[0][nt] = wmma_bf16(a0A, bA[nt], acc[0][nt]);
        acc[1][nt] = wmma_bf16(a1A, bA[nt], acc[1][nt]);
    }
#pragma unroll
    for (int nt = 0; nt < 4; ++nt) {
        acc[0][nt] = wmma_bf16(a0B, bB[nt], acc[0][nt]);
        acc[1][nt] = wmma_bf16(a1B, bB[nt], acc[1][nt]);
    }
    SCHED_GROUP(SG_WMMA, 8, 0);
    SCHED_GROUP(SG_WMMA, 8, 0);

    // D fragment: element i lives at (m = 16*r + i + 8*half, n = nt*16 + row)
#pragma unroll
    for (int r = 0; r < 2; ++r)
#pragma unroll
        for (int nt = 0; nt < 4; ++nt)
#pragma unroll
            for (int i = 0; i < 8; ++i)
                lds[wave][16 * r + i + 8 * half][nt * 16 + row] = acc[r][nt][i];

    __syncthreads();

    // ---- vectorized elementwise LSTM update: each thread = 2 x 8 contiguous cols ----
#pragma unroll
    for (int rr = 0; rr < 2; ++rr) {
        const int tid = threadIdx.x;
        const int m   = (tid >> 3) + 16 * rr;    // 0..31
        const int jb  = (tid & 7) * 8;           // 0,8,...,56
        const int b   = m0 + m;
        const int jj  = j0 + jb;

        F8 gi, gf, gg, go, co, ho;
        const float* fb = featb_l + (size_t)b * G4;
        gi.v[0] = *reinterpret_cast<const float4*>(&lds[0][m][jb]);
        gi.v[1] = *reinterpret_cast<const float4*>(&lds[0][m][jb + 4]);
        gf.v[0] = *reinterpret_cast<const float4*>(&lds[1][m][jb]);
        gf.v[1] = *reinterpret_cast<const float4*>(&lds[1][m][jb + 4]);
        gg.v[0] = *reinterpret_cast<const float4*>(&lds[2][m][jb]);
        gg.v[1] = *reinterpret_cast<const float4*>(&lds[2][m][jb + 4]);
        go.v[0] = *reinterpret_cast<const float4*>(&lds[3][m][jb]);
        go.v[1] = *reinterpret_cast<const float4*>(&lds[3][m][jb + 4]);

        F8 fi, ff, fg, fo;
        fi.v[0] = *reinterpret_cast<const float4*>(fb + 0 * HH + jj);
        fi.v[1] = *reinterpret_cast<const float4*>(fb + 0 * HH + jj + 4);
        ff.v[0] = *reinterpret_cast<const float4*>(fb + 1 * HH + jj);
        ff.v[1] = *reinterpret_cast<const float4*>(fb + 1 * HH + jj + 4);
        fg.v[0] = *reinterpret_cast<const float4*>(fb + 2 * HH + jj);
        fg.v[1] = *reinterpret_cast<const float4*>(fb + 2 * HH + jj + 4);
        fo.v[0] = *reinterpret_cast<const float4*>(fb + 3 * HH + jj);
        fo.v[1] = *reinterpret_cast<const float4*>(fb + 3 * HH + jj + 4);

        const size_t sidx = (size_t)b * HH + jj;
        co.v[0] = *reinterpret_cast<const float4*>(c_f32 + sidx);
        co.v[1] = *reinterpret_cast<const float4*>(c_f32 + sidx + 4);
        ho.v[0] = *reinterpret_cast<const float4*>(h_f32 + sidx);
        ho.v[1] = *reinterpret_cast<const float4*>(h_f32 + sidx + 4);

        const float msk = (t < length[b]) ? 1.0f : 0.0f;

        F8 hm, cm;
#pragma unroll
        for (int e = 0; e < 8; ++e) {
            float iv = gi.s[e] + fi.s[e];
            float fv = gf.s[e] + ff.s[e];
            float gv = gg.s[e] + fg.s[e];
            float ov = go.s[e] + fo.s[e];
            float si = sigmoidf_fast(iv);
            float sf = sigmoidf_fast(fv);
            float so = sigmoidf_fast(ov);
            float cn = sf * co.s[e] + si * tanhf(gv);
            float hn = so * tanhf(cn);
            hm.s[e] = fmaf(msk, hn - ho.s[e], ho.s[e]);
            cm.s[e] = fmaf(msk, cn - co.s[e], co.s[e]);
        }

        *reinterpret_cast<float4*>(h_f32 + sidx)     = hm.v[0];
        *reinterpret_cast<float4*>(h_f32 + sidx + 4) = hm.v[1];
        *reinterpret_cast<float4*>(c_f32 + sidx)     = cm.v[0];
        *reinterpret_cast<float4*>(c_f32 + sidx + 4) = cm.v[1];

        uint4 hb;
        hb.x = pack_bf2(hm.s[0], hm.s[1]);
        hb.y = pack_bf2(hm.s[2], hm.s[3]);
        hb.z = pack_bf2(hm.s[4], hm.s[5]);
        hb.w = pack_bf2(hm.s[6], hm.s[7]);
        *reinterpret_cast<uint4*>(hnext_bf + sidx) = hb;

        const size_t oidx = (size_t)t * BB * HH + sidx;
        if (yout_bf) *reinterpret_cast<uint4*>(yout_bf + oidx) = hb;
        if (yout_f32) {
            *reinterpret_cast<float4*>(yout_f32 + oidx)     = hm.v[0];
            *reinterpret_cast<float4*>(yout_f32 + oidx + 4) = hm.v[1];
        }
        if (hn_out) {
            *reinterpret_cast<float4*>(hn_out + sidx)     = hm.v[0];
            *reinterpret_cast<float4*>(hn_out + sidx + 4) = hm.v[1];
            *reinterpret_cast<float4*>(cn_out + sidx)     = cm.v[0];
            *reinterpret_cast<float4*>(cn_out + sidx + 4) = cm.v[1];
        }
    }
}

// ---------------- host orchestration ----------------
extern "C" void kernel_launch(void* const* d_in, const int* in_sizes, int n_in,
                              void* d_out, int out_size, void* d_ws, size_t ws_size,
                              hipStream_t stream) {
    (void)in_sizes; (void)n_in; (void)out_size; (void)ws_size;

    const float* x        = (const float*)d_in[0];   // (T,B,D)
    const float* features = (const float*)d_in[1];   // (B,F)
    const float* Wih      = (const float*)d_in[2];   // (L,4H,D)
    const float* Whh      = (const float*)d_in[3];   // (L,4H,H)
    const float* Wfh      = (const float*)d_in[4];   // (L,4H,F)
    const float* bias     = (const float*)d_in[5];   // (L,4H)
    const int*   length   = (const int*)d_in[6];     // (B,)
    float*       out      = (float*)d_out;           // out(T,B,H) | h_n(L,B,H) | c_n(L,B,H)

    // ---- workspace layout ----
    char* ws = (char*)d_ws;
    size_t off = 0;
    auto alloc = [&](size_t bytes) { char* p = ws + off; off += (bytes + 255) & ~(size_t)255; return p; };

    unsigned short* Wih_bf  = (unsigned short*)alloc((size_t)LL * G4 * DD * 2); // 16MB
    unsigned short* Whh_bf  = (unsigned short*)alloc((size_t)LL * G4 * HH * 2); // 16MB
    unsigned short* Wfh_bf  = (unsigned short*)alloc((size_t)LL * G4 * FF * 2); // 16MB
    unsigned short* x_bf    = (unsigned short*)alloc((size_t)TT * BB * DD * 2); // 32MB
    unsigned short* y_bf    = (unsigned short*)alloc((size_t)TT * BB * HH * 2); // 32MB
    unsigned short* feat_bf = (unsigned short*)alloc((size_t)BB * FF * 2);      // 512KB
    float*          featb   = (float*)alloc((size_t)LL * BB * G4 * 4);          // 8MB
    // state block (zeroed together each layer): h_f32 | c_f32 | hbf0 | hbf1
    float*          h_f32   = (float*)alloc((size_t)BB * HH * 4);               // 1MB
    float*          c_f32   = (float*)alloc((size_t)BB * HH * 4);               // 1MB
    unsigned short* hbf0    = (unsigned short*)alloc((size_t)BB * HH * 2);      // 512KB
    unsigned short* hbf1    = (unsigned short*)alloc((size_t)BB * HH * 2);      // 512KB

    // ---- 1) fp32 -> bf16 conversions ----
    auto cvt = [&](const float* src, unsigned short* dst, size_t n) {
        int n4 = (int)(n / 4);
        int blocks = (n4 + 255) / 256;
        if (blocks > 4096) blocks = 4096;
        cvt_f32_bf16_kernel<<<blocks, 256, 0, stream>>>((const float4*)src, (ushort4*)dst, n4);
    };
    cvt(Wih,      Wih_bf,  (size_t)LL * G4 * DD);
    cvt(Whh,      Whh_bf,  (size_t)LL * G4 * HH);
    cvt(Wfh,      Wfh_bf,  (size_t)LL * G4 * FF);
    cvt(x,        x_bf,    (size_t)TT * BB * DD);
    cvt(features, feat_bf, (size_t)BB * FF);

    // ---- 2) featb[l] = features @ Wfh[l]^T + b[l] ----
    {
        dim3 grid(G4 / 64, BB / 64, LL);
        featb_gemm_kernel<<<grid, 128, 0, stream>>>(feat_bf, Wfh_bf, bias, featb);
    }

    // ---- 3) recurrent layers ----
    float* hn_base = out + (size_t)TT * BB * HH;
    float* cn_base = hn_base + (size_t)LL * BB * HH;

    for (int l = 0; l < LL; ++l) {
        // zero h_f32 | c_f32 | hbf0 | hbf1 (contiguous, 3MB -> 786432 u32)
        {
            int nwords = (int)(((size_t)BB * HH * 4 * 2 + (size_t)BB * HH * 2 * 2) / 4);
            zero_u32_kernel<<<1024, 256, 0, stream>>>((unsigned*)h_f32, nwords);
        }
        const unsigned short* xin  = (l == 0) ? x_bf : y_bf;
        const unsigned short* Wihl = Wih_bf + (size_t)l * G4 * DD;
        const unsigned short* Whhl = Whh_bf + (size_t)l * G4 * HH;
        const float*          fbl  = featb  + (size_t)l * BB * G4;
        unsigned short* ybf  = (l == 0) ? y_bf : nullptr;
        float*          yf32 = (l == LL - 1) ? out : nullptr;

        dim3 grid(HH / 64, BB / 32);
        for (int t = 0; t < TT; ++t) {
            unsigned short* hprev = (t & 1) ? hbf1 : hbf0;
            unsigned short* hnext = (t & 1) ? hbf0 : hbf1;
            float* hn = (t == TT - 1) ? (hn_base + (size_t)l * BB * HH) : nullptr;
            float* cn = (t == TT - 1) ? (cn_base + (size_t)l * BB * HH) : nullptr;
            lstm_step_kernel<<<grid, 128, 0, stream>>>(
                xin, hprev, hnext, Wihl, Whhl, fbl, length,
                h_f32, c_f32, ybf, yf32, hn, cn, t);
        }
    }
}
